// TemporalClassificationModel_51745765982421
// MI455X (gfx1250) — matrix operand order repaired and checked by reading, not compiled, not run on hardware
//
#include <hip/hip_runtime.h>

typedef __attribute__((ext_vector_type(16))) __bf16 v16bf;
typedef __attribute__((ext_vector_type(8)))  __bf16 v8bf;
typedef __attribute__((ext_vector_type(4)))  __bf16 v4bf;
typedef __attribute__((ext_vector_type(8)))  float  v8f;

#define BB 64
#define LL 512
#define DD 768
#define TT 12
#define BM 128
#define BN 128
#define BK 32
#define KP 40   // padded LDS K stride (elements): 80B row pitch -> conflict-free 16-lane access

union Frag { v16bf v; v8bf h[2]; };

// ---------------- bitmap for isin() ----------------
__global__ void build_bitmap(const int* __restrict__ vf, int F,
                             unsigned* __restrict__ bitmap) {
  int t = threadIdx.x;
  bitmap[t] = 0u;                 // 1024 words cover 30522 bits
  __syncthreads();
  if (t < F) {
    int v = vf[t];
    atomicOr(&bitmap[v >> 5], 1u << (v & 31));
  }
}

// ---------------- W1/W2 f32 -> bf16 ----------------
__global__ void convert_weights(const float* __restrict__ W1,
                                const float* __restrict__ W2,
                                __bf16* __restrict__ W1b,
                                __bf16* __restrict__ W2b) {
  size_t i4 = (size_t)blockIdx.x * blockDim.x + threadIdx.x;
  size_t n4 = (size_t)TT * DD * DD / 4;
  if (i4 >= n4) return;
  float4 a = reinterpret_cast<const float4*>(W1)[i4];
  float4 b = reinterpret_cast<const float4*>(W2)[i4];
  v4bf av = { (__bf16)a.x, (__bf16)a.y, (__bf16)a.z, (__bf16)a.w };
  v4bf bv = { (__bf16)b.x, (__bf16)b.y, (__bf16)b.z, (__bf16)b.w };
  *reinterpret_cast<v4bf*>(W1b + i4 * 4) = av;
  *reinterpret_cast<v4bf*>(W2b + i4 * 4) = bv;
}

// -------- gather embeddings + bf16 copy + mask --------
__global__ void gather_embs(const int* __restrict__ ids,
                            const float* __restrict__ emb,
                            float* __restrict__ bert,      // d_out part 1
                            __bf16* __restrict__ Xb,
                            const unsigned* __restrict__ bitmap,
                            float* __restrict__ maskf) {
  int i = blockIdx.x;            // token index [0, B*L)
  int t = threadIdx.x;           // 0..191 (DD/4)
  int id = ids[i];
  float4 v = reinterpret_cast<const float4*>(emb + (size_t)id * DD)[t];
  reinterpret_cast<float4*>(bert + (size_t)i * DD)[t] = v;
  v4bf o = { (__bf16)v.x, (__bf16)v.y, (__bf16)v.z, (__bf16)v.w };
  *reinterpret_cast<v4bf*>(Xb + (size_t)i * DD + t * 4) = o;
  if (t == 0) {
    unsigned w = bitmap[id >> 5];
    maskf[i] = (float)((w >> (id & 31)) & 1u);
  }
}

// ---------------- shared WMMA GEMM core ----------------
// acc[mt][nt] += A[128xK] * W[128xK]^T tile; A rows = tokens, W rows = output features.
__device__ __forceinline__ void gemm_core(const __bf16* __restrict__ Ag,
                                          const __bf16* __restrict__ Bg,
                                          __bf16 (*Asl)[KP],
                                          __bf16 (*Bsl)[KP],
                                          v8f acc[2][4]) {
  const int tid  = threadIdx.x;
  const int lane = tid & 31;
  const int wave = tid >> 5;
  const int wm   = wave & 3;       // 4 waves along M (32 rows each)
  const int wn   = wave >> 2;      // 2 waves along N (64 cols each)
  const int l16  = lane & 15;
  const int half = lane >> 4;

  // cooperative tile load mapping: 512 16B chunks per tile, 2 per thread
  const int r0 = tid >> 2,          k0c = (tid & 3) * 8;
  const int r1 = (tid + 256) >> 2,  k1c = ((tid + 256) & 3) * 8;

  for (int k0 = 0; k0 < DD; k0 += BK) {
    __syncthreads();
    *reinterpret_cast<int4*>(&Asl[r0][k0c]) =
        *reinterpret_cast<const int4*>(Ag + (size_t)r0 * DD + k0 + k0c);
    *reinterpret_cast<int4*>(&Asl[r1][k1c]) =
        *reinterpret_cast<const int4*>(Ag + (size_t)r1 * DD + k0 + k1c);
    *reinterpret_cast<int4*>(&Bsl[r0][k0c]) =
        *reinterpret_cast<const int4*>(Bg + (size_t)r0 * DD + k0 + k0c);
    *reinterpret_cast<int4*>(&Bsl[r1][k1c]) =
        *reinterpret_cast<const int4*>(Bg + (size_t)r1 * DD + k0 + k1c);
    __syncthreads();

    // A frags: lanes 0-15 hold K chunks {0-7,16-23}, lanes 16-31 {8-15,24-31}
    Frag af[2];
#pragma unroll
    for (int mt = 0; mt < 2; ++mt) {
      int row = wm * 32 + mt * 16 + l16;
      int kb  = half * 8;
      af[mt].h[0] = *reinterpret_cast<const v8bf*>(&Asl[row][kb]);
      af[mt].h[1] = *reinterpret_cast<const v8bf*>(&Asl[row][kb + 16]);
    }
    // B frags: lane holds column (= W row), K chunks {0-7,16-23} (lanes 0-15) / {8-15,24-31} (16-31)
#pragma unroll
    for (int nt = 0; nt < 4; ++nt) {
      Frag bfg;
      int col = wn * 64 + nt * 16 + l16;
      int kb  = half * 8;
      bfg.h[0] = *reinterpret_cast<const v8bf*>(&Bsl[col][kb]);
      bfg.h[1] = *reinterpret_cast<const v8bf*>(&Bsl[col][kb + 16]);
#pragma unroll
      for (int mt = 0; mt < 2; ++mt) {
        acc[mt][nt] = __builtin_amdgcn_wmma_f32_16x16x32_bf16(
            false, af[mt].v, false, bfg.v, (short)0, acc[mt][nt], false, false);
      }
    }
  }
}

// ---------------- GEMM1: H = tanh(X W1^T + b1) ----------------
__global__ __launch_bounds__(256) void gemm1_kernel(
    const __bf16* __restrict__ Xb, const __bf16* __restrict__ W1b,
    const float* __restrict__ b1, const int* __restrict__ timediff,
    __bf16* __restrict__ Hb) {
  __shared__ __align__(16) __bf16 Asl[BM][KP];
  __shared__ __align__(16) __bf16 Bsl[BM][KP];
  const int b = blockIdx.z, m0 = blockIdx.y * BM, n0 = blockIdx.x * BN;
  const int t = timediff[b];
  const __bf16* Ag = Xb + ((size_t)b * LL + m0) * DD;
  const __bf16* Bg = W1b + (size_t)t * DD * DD + (size_t)n0 * DD;

  v8f zero = {0.f,0.f,0.f,0.f,0.f,0.f,0.f,0.f};
  v8f acc[2][4];
#pragma unroll
  for (int mt = 0; mt < 2; ++mt)
#pragma unroll
    for (int nt = 0; nt < 4; ++nt) acc[mt][nt] = zero;

  gemm_core(Ag, Bg, Asl, Bsl, acc);

  const int lane = threadIdx.x & 31, wave = threadIdx.x >> 5;
  const int wm = wave & 3, wn = wave >> 2;
  const int l16 = lane & 15, half = lane >> 4;
#pragma unroll
  for (int nt = 0; nt < 4; ++nt) {
    int n = n0 + wn * 64 + nt * 16 + l16;
    float bias = b1[t * DD + n];
#pragma unroll
    for (int mt = 0; mt < 2; ++mt) {
      int mbase = m0 + wm * 32 + mt * 16 + half * 8;
#pragma unroll
      for (int r = 0; r < 8; ++r) {
        float hv = tanhf(acc[mt][nt][r] + bias);
        Hb[((size_t)b * LL + mbase + r) * DD + n] = (__bf16)hv;
      }
    }
  }
}

// -------- GEMM2: out2 = bert + (H W2^T + b2) * mask --------
__global__ __launch_bounds__(256) void gemm2_kernel(
    const __bf16* __restrict__ Hb, const __bf16* __restrict__ W2b,
    const float* __restrict__ b2, const int* __restrict__ timediff,
    const float* __restrict__ maskf, const float* __restrict__ bert,
    float* __restrict__ out2) {
  __shared__ __align__(16) __bf16 Asl[BM][KP];
  __shared__ __align__(16) __bf16 Bsl[BM][KP];
  const int b = blockIdx.z, m0 = blockIdx.y * BM, n0 = blockIdx.x * BN;
  const int t = timediff[b];
  const __bf16* Ag = Hb + ((size_t)b * LL + m0) * DD;
  const __bf16* Bg = W2b + (size_t)t * DD * DD + (size_t)n0 * DD;

  v8f zero = {0.f,0.f,0.f,0.f,0.f,0.f,0.f,0.f};
  v8f acc[2][4];
#pragma unroll
  for (int mt = 0; mt < 2; ++mt)
#pragma unroll
    for (int nt = 0; nt < 4; ++nt) acc[mt][nt] = zero;

  gemm_core(Ag, Bg, Asl, Bsl, acc);

  const int lane = threadIdx.x & 31, wave = threadIdx.x >> 5;
  const int wm = wave & 3, wn = wave >> 2;
  const int l16 = lane & 15, half = lane >> 4;
#pragma unroll
  for (int nt = 0; nt < 4; ++nt) {
    int n = n0 + wn * 64 + nt * 16 + l16;
    float bias = b2[t * DD + n];
#pragma unroll
    for (int mt = 0; mt < 2; ++mt) {
      int mbase = m0 + wm * 32 + mt * 16 + half * 8;
#pragma unroll
      for (int r = 0; r < 8; ++r) {
        int m = mbase + r;
        size_t idx = ((size_t)b * LL + m) * DD + n;
        float off = acc[mt][nt][r] + bias;
        out2[idx] = bert[idx] + off * maskf[b * LL + m];
      }
    }
  }
}

extern "C" void kernel_launch(void* const* d_in, const int* in_sizes, int n_in,
                              void* d_out, int out_size, void* d_ws, size_t ws_size,
                              hipStream_t stream) {
  const int*   input_ids = (const int*)d_in[0];
  const int*   timediff  = (const int*)d_in[1];
  const int*   vfilt     = (const int*)d_in[2];
  const float* emb       = (const float*)d_in[3];
  const float* W1        = (const float*)d_in[4];
  const float* b1        = (const float*)d_in[5];
  const float* W2        = (const float*)d_in[6];
  const float* b2        = (const float*)d_in[7];

  float* out1 = (float*)d_out;                       // bert_embs
  float* out2 = out1 + (size_t)BB * LL * DD;         // input_embs

  char* ws = (char*)d_ws;
  __bf16*  Xb     = (__bf16*)ws;  ws += (size_t)BB * LL * DD * 2;
  __bf16*  Hb     = (__bf16*)ws;  ws += (size_t)BB * LL * DD * 2;
  __bf16*  W1b    = (__bf16*)ws;  ws += (size_t)TT * DD * DD * 2;
  __bf16*  W2b    = (__bf16*)ws;  ws += (size_t)TT * DD * DD * 2;
  float*   maskf  = (float*)ws;   ws += (size_t)BB * LL * 4;
  unsigned* bitmap = (unsigned*)ws;

  build_bitmap<<<1, 1024, 0, stream>>>(vfilt, in_sizes[2], bitmap);

  size_t n4 = (size_t)TT * DD * DD / 4;
  convert_weights<<<(unsigned)((n4 + 255) / 256), 256, 0, stream>>>(W1, W2, W1b, W2b);

  gather_embs<<<BB * LL, DD / 4, 0, stream>>>(input_ids, emb, out1, Xb, bitmap, maskf);

  dim3 gg(DD / BN, LL / BM, BB);
  gemm1_kernel<<<gg, 256, 0, stream>>>(Xb, W1b, b1, timediff, Hb);
  gemm2_kernel<<<gg, 256, 0, stream>>>(Hb, W2b, b2, timediff, maskf, out1, out2);
}
